// FlowSAN_54838142435870
// MI455X (gfx1250) — matrix operand order repaired and checked by reading, not compiled, not run on hardware
//
#include <hip/hip_runtime.h>
#include <math.h>

#define N_EDGES 200000
#define NNZ_CNT 2000000
#define GBATCH  256
#define NEG_SLOPE 0.2f

typedef __attribute__((ext_vector_type(16))) _Float16 v16h;
typedef __attribute__((ext_vector_type(8)))  float    v8f;

// ---------------- utilities ----------------
__global__ void fill_u32_kernel(unsigned* __restrict__ p, unsigned v, int n) {
    int i = blockIdx.x * blockDim.x + threadIdx.x;
    if (i < n) p[i] = v;
}

// Monotonic float<->int encoding so signed-int atomicMax implements float max.
__device__ __forceinline__ int enc_f32(float x) {
    int b = __float_as_int(x);
    return (b >= 0) ? b : (b ^ 0x7fffffff);
}
__device__ __forceinline__ float dec_f32(int e) {
    int b = (e >= 0) ? e : (e ^ 0x7fffffff);
    return __int_as_float(b);
}
#define ENC_NEG_INF 0x807fffffu   // enc(-inf)

__device__ __forceinline__ float lrelu(float t) { return t > 0.f ? t : NEG_SLOPE * t; }

// ---------------- dense GEMM via WMMA (H = X[fi] @ W[fi x fo]) ----------------
// One wave32 computes one 16x16 output tile; K-loop in steps of 32 using
// v_wmma_f32_16x16x32_f16. fp32 inputs converted to f16 on load (GEMM is
// ~0.1% of total work; matrix-core path is effectively free).
template<int FI, int FO>
__global__ __launch_bounds__(256) void gemm_wmma_kernel(const float* __restrict__ X,
                                                        const float* __restrict__ W,
                                                        float* __restrict__ H) {
    constexpr int TN = (FO + 15) / 16;
    const int tilesM = N_EDGES / 16;
    const int wave = blockIdx.x * (blockDim.x >> 5) + (threadIdx.x >> 5);
    if (wave >= tilesM * TN) return;                 // whole-wave uniform exit: EXEC stays all-1s
    const int mt = wave / TN;
    const int nt = wave % TN;
    const int lane = threadIdx.x & 31;
    const int half = lane >> 4;                      // lane group 0: lanes 0-15, 1: lanes 16-31
    const int l16  = lane & 15;

    const int rowA = mt * 16 + l16;                  // A: M = l16 in both halves
    const int colB = nt * 16 + l16;                  // B: N = l16 in both halves
    v8f c = {};

    for (int kk = 0; kk < FI; kk += 32) {
        v16h a, b;
        const float* xr = X + (size_t)rowA * FI + kk;
        #pragma unroll
        for (int j = 0; j < 16; ++j) {
            // A k order chosen here (contiguous, same as B): lanes0-15 K=0..15, lanes16-31 K=16..31
            int k = half * 16 + j;
            a[j] = (_Float16)xr[k];
        }
        #pragma unroll
        for (int j = 0; j < 16; ++j) {
            // B k order chosen here (contiguous, same as A): lanes0-15 K=0..15, lanes16-31 K=16..31
            int k = kk + half * 16 + j;
            float wv = (colB < FO) ? W[(size_t)k * FO + colB] : 0.f;
            b[j] = (_Float16)wv;
        }
        c = __builtin_amdgcn_wmma_f32_16x16x32_f16(false, a, false, b, (short)0, c, false, false);
    }
    #pragma unroll
    for (int r = 0; r < 8; ++r) {
        int row = mt * 16 + half * 8 + r;            // D: M = r (+8 for upper lanes)
        if (colB < FO) H[(size_t)row * FO + colB] = c[r];
    }
}

// Per-row attention scalars: s = h . a  (collapses per-edge dot products to scalar gathers)
template<int FO>
__global__ void rowdots_kernel(const float* __restrict__ hu, const float* __restrict__ hd,
                               const float* __restrict__ uas, const float* __restrict__ uad,
                               const float* __restrict__ das, const float* __restrict__ dad,
                               float* __restrict__ sus, float* __restrict__ sud,
                               float* __restrict__ sds, float* __restrict__ sdd) {
    int i = blockIdx.x * blockDim.x + threadIdx.x;
    if (i >= N_EDGES) return;
    const float* ru = hu + (size_t)i * FO;
    const float* rd = hd + (size_t)i * FO;
    float a0 = 0.f, a1 = 0.f, a2 = 0.f, a3 = 0.f;
    #pragma unroll
    for (int f = 0; f < FO; ++f) {
        a0 += ru[f] * uas[f]; a1 += ru[f] * uad[f];
        a2 += rd[f] * das[f]; a3 += rd[f] * dad[f];
    }
    sus[i] = a0; sud[i] = a1; sds[i] = a2; sdd[i] = a3;
}

// p-branch: acc[row] += l1_val * hp[col].  Grid over (nnz x fo): consecutive
// lanes cover one edge's features -> coalesced reads + coalesced atomics.
template<int FO>
__global__ void pscatter_kernel(const int* __restrict__ idx, const float* __restrict__ val,
                                const float* __restrict__ hp, float* __restrict__ acc) {
    constexpr int LF = (FO == 8) ? 3 : 5;
    int t = blockIdx.x * blockDim.x + threadIdx.x;      // NNZ*FO <= 64M < 2^31
    if (t >= NNZ_CNT * FO) return;
    int f = t & (FO - 1);
    int e = t >> LF;
    int row = idx[e];
    int col = idx[NNZ_CNT + e];
    atomicAdd(&acc[(size_t)row * FO + f], val[e] * hp[(size_t)col * FO + f]);
}

// GAT pass A: per-edge logit -> segment max (float max via int-encoded atomicMax)
__global__ void attmax_kernel(const int* __restrict__ idx,
                              const float* __restrict__ ss, const float* __restrict__ sd,
                              int* __restrict__ rmax) {
    int e = blockIdx.x * blockDim.x + threadIdx.x;
    if (e >= NNZ_CNT) return;
    int row = idx[e], col = idx[NNZ_CNT + e];
    float ev = lrelu(ss[row] + sd[col]);
    atomicMax(&rmax[row], enc_f32(ev));
}

// GAT pass B: denominators
__global__ void attden_kernel(const int* __restrict__ idx,
                              const float* __restrict__ ss, const float* __restrict__ sd,
                              const int* __restrict__ rmax, float* __restrict__ den) {
    int e = blockIdx.x * blockDim.x + threadIdx.x;
    if (e >= NNZ_CNT) return;
    int row = idx[e], col = idx[NNZ_CNT + e];
    float ev = lrelu(ss[row] + sd[col]);
    atomicAdd(&den[row], __expf(ev - dec_f32(rmax[row])));
}

// GAT pass C: acc[row] += alpha * h[col], alpha recomputed (cheap VALU, saves 16MB of w-storage)
template<int FO>
__global__ void attscatter_kernel(const int* __restrict__ idx,
                                  const float* __restrict__ ss, const float* __restrict__ sd,
                                  const int* __restrict__ rmax, const float* __restrict__ den,
                                  const float* __restrict__ h, float* __restrict__ acc) {
    constexpr int LF = (FO == 8) ? 3 : 5;
    int t = blockIdx.x * blockDim.x + threadIdx.x;
    if (t >= NNZ_CNT * FO) return;
    int f = t & (FO - 1);
    int e = t >> LF;
    int row = idx[e], col = idx[NNZ_CNT + e];
    float ev = lrelu(ss[row] + sd[col]);
    float alpha = __expf(ev - dec_f32(rmax[row])) / (den[row] + 1e-16f);
    atomicAdd(&acc[(size_t)row * FO + f], alpha * h[(size_t)col * FO + f]);
}

__global__ void relu_kernel(const float* __restrict__ a, float* __restrict__ o, int n) {
    int i = blockIdx.x * blockDim.x + threadIdx.x;
    if (i < n) o[i] = fmaxf(a[i], 0.f);
}

__global__ void pool_kernel(const float* __restrict__ x, const int* __restrict__ batch,
                            float* __restrict__ pool, float* __restrict__ cnt) {
    int i = blockIdx.x * blockDim.x + threadIdx.x;
    if (i >= N_EDGES) return;
    int b = batch[i];
    atomicAdd(&cnt[b], 1.0f);
    #pragma unroll
    for (int f = 0; f < 8; ++f)
        atomicAdd(&pool[b * 8 + f], fabsf(x[(size_t)i * 8 + f]));
}

__global__ void softmax_kernel(const float* __restrict__ pool, const float* __restrict__ cnt,
                               float* __restrict__ out) {
    int g = blockIdx.x * blockDim.x + threadIdx.x;
    if (g >= GBATCH) return;
    float c = fmaxf(cnt[g], 1.0f);
    float v[8]; float m = -1e30f;
    #pragma unroll
    for (int f = 0; f < 8; ++f) { v[f] = pool[g * 8 + f] / c; m = fmaxf(m, v[f]); }
    float s = 0.f;
    #pragma unroll
    for (int f = 0; f < 8; ++f) { v[f] = __expf(v[f] - m); s += v[f]; }
    #pragma unroll
    for (int f = 0; f < 8; ++f) out[g * 8 + f] = v[f] / s;
}

// ---------------- host-side orchestration ----------------
static inline int cdiv(long long a, long long b) { return (int)((a + b - 1) / b); }

template<int FI, int FO>
static void run_layer(const float* X,
                      const float* pW, const float* uW, const float* uas, const float* uad,
                      const float* dW, const float* das, const float* dad,
                      const int* l1i, const float* l1v, const int* lui, const int* ldi,
                      float* hp, float* hu, float* hd, float* acc, float* xout,
                      float* sus, float* sud, float* sds, float* sdd,
                      int* rmu, int* rmd, float* denu, float* dend,
                      hipStream_t s) {
    fill_u32_kernel<<<cdiv(N_EDGES * FO, 256), 256, 0, s>>>((unsigned*)acc, 0u, N_EDGES * FO);
    fill_u32_kernel<<<cdiv(N_EDGES, 256), 256, 0, s>>>((unsigned*)rmu, ENC_NEG_INF, N_EDGES);
    fill_u32_kernel<<<cdiv(N_EDGES, 256), 256, 0, s>>>((unsigned*)rmd, ENC_NEG_INF, N_EDGES);
    fill_u32_kernel<<<cdiv(N_EDGES, 256), 256, 0, s>>>((unsigned*)denu, 0u, N_EDGES);
    fill_u32_kernel<<<cdiv(N_EDGES, 256), 256, 0, s>>>((unsigned*)dend, 0u, N_EDGES);

    constexpr int TN = (FO + 15) / 16;
    int gemm_blocks = cdiv((long long)(N_EDGES / 16) * TN, 8);
    gemm_wmma_kernel<FI, FO><<<gemm_blocks, 256, 0, s>>>(X, pW, hp);
    gemm_wmma_kernel<FI, FO><<<gemm_blocks, 256, 0, s>>>(X, uW, hu);
    gemm_wmma_kernel<FI, FO><<<gemm_blocks, 256, 0, s>>>(X, dW, hd);

    rowdots_kernel<FO><<<cdiv(N_EDGES, 256), 256, 0, s>>>(hu, hd, uas, uad, das, dad,
                                                          sus, sud, sds, sdd);

    pscatter_kernel<FO><<<cdiv((long long)NNZ_CNT * FO, 256), 256, 0, s>>>(l1i, l1v, hp, acc);

    attmax_kernel<<<cdiv(NNZ_CNT, 256), 256, 0, s>>>(lui, sus, sud, rmu);
    attmax_kernel<<<cdiv(NNZ_CNT, 256), 256, 0, s>>>(ldi, sds, sdd, rmd);
    attden_kernel<<<cdiv(NNZ_CNT, 256), 256, 0, s>>>(lui, sus, sud, rmu, denu);
    attden_kernel<<<cdiv(NNZ_CNT, 256), 256, 0, s>>>(ldi, sds, sdd, rmd, dend);
    attscatter_kernel<FO><<<cdiv((long long)NNZ_CNT * FO, 256), 256, 0, s>>>(lui, sus, sud, rmu, denu, hu, acc);
    attscatter_kernel<FO><<<cdiv((long long)NNZ_CNT * FO, 256), 256, 0, s>>>(ldi, sds, sdd, rmd, dend, hd, acc);

    relu_kernel<<<cdiv(N_EDGES * FO, 256), 256, 0, s>>>(acc, xout, N_EDGES * FO);
}

extern "C" void kernel_launch(void* const* d_in, const int* in_sizes, int n_in,
                              void* d_out, int out_size, void* d_ws, size_t ws_size,
                              hipStream_t stream) {
    (void)in_sizes; (void)n_in; (void)out_size; (void)ws_size;
    const float* x1   = (const float*)d_in[0];
    // layer param block: d_in[1 + 7*l + {0..6}] = pW,uW,ua_s,ua_d,dW,da_s,da_d
    const int*   l1i  = (const int*)  d_in[29];
    const float* l1v  = (const float*)d_in[30];
    const int*   lui  = (const int*)  d_in[31];
    const int*   ldi  = (const int*)  d_in[32];
    const int*   bat  = (const int*)  d_in[33];
    float* out = (float*)d_out;

    // workspace carve (floats): 5 * N*32 feature buffers + 8 * N scalars + pool
    float* ws   = (float*)d_ws;
    float* hp   = ws;
    float* hu   = hp + (size_t)N_EDGES * 32;
    float* hd   = hu + (size_t)N_EDGES * 32;
    float* acc  = hd + (size_t)N_EDGES * 32;
    float* xb   = acc + (size_t)N_EDGES * 32;
    float* sus  = xb + (size_t)N_EDGES * 32;
    float* sud  = sus + N_EDGES;
    float* sds  = sud + N_EDGES;
    float* sdd  = sds + N_EDGES;
    int*   rmu  = (int*)(sdd + N_EDGES);
    int*   rmd  = rmu + N_EDGES;
    float* denu = (float*)(rmd + N_EDGES);
    float* dend = denu + N_EDGES;
    float* pool = dend + N_EDGES;          // GBATCH*8
    float* cnt  = pool + GBATCH * 8;       // GBATCH

    #define LP(l, k) ((const float*)d_in[1 + 7 * (l) + (k)])

    run_layer<64, 32>(x1, LP(0,0), LP(0,1), LP(0,2), LP(0,3), LP(0,4), LP(0,5), LP(0,6),
                      l1i, l1v, lui, ldi, hp, hu, hd, acc, xb,
                      sus, sud, sds, sdd, rmu, rmd, denu, dend, stream);
    run_layer<32, 32>(xb, LP(1,0), LP(1,1), LP(1,2), LP(1,3), LP(1,4), LP(1,5), LP(1,6),
                      l1i, l1v, lui, ldi, hp, hu, hd, acc, xb,
                      sus, sud, sds, sdd, rmu, rmd, denu, dend, stream);
    run_layer<32, 32>(xb, LP(2,0), LP(2,1), LP(2,2), LP(2,3), LP(2,4), LP(2,5), LP(2,6),
                      l1i, l1v, lui, ldi, hp, hu, hd, acc, xb,
                      sus, sud, sds, sdd, rmu, rmd, denu, dend, stream);
    run_layer<32, 8>(xb, LP(3,0), LP(3,1), LP(3,2), LP(3,3), LP(3,4), LP(3,5), LP(3,6),
                     l1i, l1v, lui, ldi, hp, hu, hd, acc, xb,
                     sus, sud, sds, sdd, rmu, rmd, denu, dend, stream);

    fill_u32_kernel<<<cdiv(GBATCH * 9, 256), 256, 0, stream>>>((unsigned*)pool, 0u, GBATCH * 9);
    pool_kernel<<<cdiv(N_EDGES, 256), 256, 0, stream>>>(xb, bat, pool, cnt);
    softmax_kernel<<<1, 256, 0, stream>>>(pool, cnt, out);
}
